// MixtureofExpertsBlock_30382598652527
// MI455X (gfx1250) — matrix operand order repaired and checked by reading, not compiled, not run on hardware
//
#include <hip/hip_runtime.h>
#include <hip/hip_bf16.h>
#include <math.h>

typedef __attribute__((ext_vector_type(16))) __bf16 v16bf;
typedef __attribute__((ext_vector_type(8)))  float  v8f;
typedef __attribute__((ext_vector_type(4)))  __bf16 v4bf;

// Problem constants (reference-fixed): B=4,S=1024 -> T tokens
#define TOK   4096
#define DDIM  1024
#define UDIM  4096
#define NEXP  8
#define NASG  (2*TOK)   // top-2 assignments

// ---------------- fp32 -> bf16 convert (vectorized) ----------------
__global__ void cvt_bf16_kernel(const float* __restrict__ src,
                                __bf16* __restrict__ dst, long n4) {
  long i = (long)blockIdx.x * blockDim.x + threadIdx.x;
  long stride = (long)gridDim.x * blockDim.x;
  for (; i < n4; i += stride) {
    float4 f = reinterpret_cast<const float4*>(src)[i];
    v4bf o;
    o[0] = (__bf16)f.x; o[1] = (__bf16)f.y;
    o[2] = (__bf16)f.z; o[3] = (__bf16)f.w;
    reinterpret_cast<v4bf*>(dst)[i] = o;
  }
}

// ---------------- router: one wave32 per token ----------------
__global__ void router_kernel(const float* __restrict__ x,
                              const float* __restrict__ wr,
                              float* __restrict__ gate,
                              int* __restrict__ counts,
                              int* __restrict__ list) {
  const int wid  = threadIdx.x >> 5;
  const int lane = threadIdx.x & 31;
  const int t = blockIdx.x * 8 + wid;
  if (t >= TOK) return;
  const float* xr = x + (long)t * DDIM;
  float acc[NEXP];
#pragma unroll
  for (int e = 0; e < NEXP; ++e) acc[e] = 0.f;
  for (int d = lane; d < DDIM; d += 32) {
    float xv = xr[d];
#pragma unroll
    for (int e = 0; e < NEXP; ++e) acc[e] += xv * wr[e * DDIM + d];
  }
#pragma unroll
  for (int off = 16; off >= 1; off >>= 1) {
#pragma unroll
    for (int e = 0; e < NEXP; ++e) acc[e] += __shfl_xor(acc[e], off, 32);
  }
  if (lane == 0) {
    float v0 = -INFINITY, v1 = -INFINITY;
    int i0 = 0, i1 = 0;
#pragma unroll
    for (int e = 0; e < NEXP; ++e) {
      float v = acc[e];
      if (v > v0)      { v1 = v0; i1 = i0; v0 = v; i0 = e; }
      else if (v > v1) { v1 = v;  i1 = e; }
    }
    float ew  = expf(v1 - v0);
    float inv = 1.0f / (1.0f + ew);
    gate[2 * t]     = inv;        // weight for top-1 assignment
    gate[2 * t + 1] = ew * inv;   // weight for top-2 assignment
    int p0 = atomicAdd(&counts[i0], 1);
    list[i0 * TOK + p0] = 2 * t;
    int p1 = atomicAdd(&counts[i1], 1);
    list[i1 * TOK + p1] = 2 * t + 1;
  }
}

// ---------------- per-expert GEMM, LDS double-buffered ----------------
// Block tile 128x128, K-step 64, 8 waves as 4(M) x 2(N), wave tile 32x64.
// LDS rows padded to 72 bf16 (144B = 36 banks) -> conflict-free b128 frag reads.
#define KSTEP 64
#define LDT   72
#define SMEM_BYTES (512 + 4 * 128 * LDT * 2)  // asg + 2x(A,B) tiles

template <int KDIM, bool IS_UP>
__global__ __launch_bounds__(256)
void moe_gemm_kernel(const __bf16* __restrict__ Abase,  // x_bf16 (up) or h_bf16 (down)
                     const __bf16* __restrict__ W,      // [E][N][K] bf16
                     const float*  __restrict__ bias,   // [E][N]
                     const float*  __restrict__ gate,   // [NASG]
                     const int*    __restrict__ counts, // [E]
                     const int*    __restrict__ list,   // [E][TOK] assignment ids
                     __bf16* __restrict__ Hout,         // h_bf16 [NASG][UDIM] (up)
                     float*  __restrict__ Dout)         // down_out [NASG][DDIM] (down)
{
  constexpr int NDIM  = IS_UP ? UDIM : DDIM;
  constexpr int STEPS = KDIM / KSTEP;
  const int e = blockIdx.z;
  const int cnt = counts[e];
  const int mtile = blockIdx.y;
  if (mtile * 128 >= cnt) return;          // uniform exit: EXEC stays all-ones
  const int ntile = blockIdx.x;

  extern __shared__ char smem[];
  int*    asg = (int*)smem;                       // 128 ints
  __bf16* Asm = (__bf16*)(smem + 512);            // 2 buffers x 128*LDT
  __bf16* Bsm = Asm + 2 * 128 * LDT;              // 2 buffers x 128*LDT

  const int tid = threadIdx.x;
  if (tid < 128) {
    int pos = mtile * 128 + tid;
    asg[tid] = list[e * TOK + (pos < cnt ? pos : cnt - 1)];
  }
  __syncthreads();

  // ---- staging: thread -> (row = tid>>1, 64B contiguous half-row) ----
  const int srow = tid >> 1;
  const int soff = (tid & 1) * 32;  // element offset within 64-elem K tile row
  const int arow_g = asg[srow];
  const __bf16* agp = IS_UP ? (Abase + (long)(arow_g >> 1) * KDIM + soff)
                            : (Abase + (long)arow_g * KDIM + soff);
  const __bf16* bgp = W + (long)e * NDIM * KDIM
                        + ((long)ntile * 128 + srow) * KDIM + soff;
  const int sst = srow * LDT + soff;  // LDS staging offset (elements)

  uint4 pa[4], pb[4];
#pragma unroll
  for (int c = 0; c < 4; ++c) {
    pa[c] = *reinterpret_cast<const uint4*>(agp + c * 8);
    pb[c] = *reinterpret_cast<const uint4*>(bgp + c * 8);
  }
#pragma unroll
  for (int c = 0; c < 4; ++c) {
    *reinterpret_cast<uint4*>(Asm + sst + c * 8) = pa[c];
    *reinterpret_cast<uint4*>(Bsm + sst + c * 8) = pb[c];
  }
  __syncthreads();

  // ---- wave/fragment coordinates ----
  const int lane  = tid & 31;
  const int wid   = tid >> 5;
  const int wm    = wid & 3;     // 0..3 : 32 rows each
  const int wn    = wid >> 2;    // 0..1 : 64 cols each
  const int lhalf = (lane >> 4) & 1;
  const int lmod  = lane & 15;
  const int mwbase = wm * 32;
  const int nwbase = wn * 64;

  v8f acc[2][4];
#pragma unroll
  for (int mf = 0; mf < 2; ++mf)
#pragma unroll
    for (int j = 0; j < 4; ++j)
#pragma unroll
      for (int r = 0; r < 8; ++r) acc[mf][j][r] = 0.f;

  union Frag { uint4 q[2]; v16bf v; };

  for (int s = 0; s < STEPS; ++s) {
    const int cur = s & 1;
    const bool more = (s + 1) < STEPS;
    if (more) {
      const long k0 = (long)(s + 1) * KSTEP;
#pragma unroll
      for (int c = 0; c < 4; ++c) {
        pa[c] = *reinterpret_cast<const uint4*>(agp + k0 + c * 8);
        pb[c] = *reinterpret_cast<const uint4*>(bgp + k0 + c * 8);
      }
      if (s + 2 < STEPS) {  // global_prefetch_b8 two K-steps ahead
        __builtin_prefetch(agp + (long)(s + 2) * KSTEP, 0, 1);
        __builtin_prefetch(bgp + (long)(s + 2) * KSTEP, 0, 1);
      }
    }
    const __bf16* Ab = Asm + cur * (128 * LDT);
    const __bf16* Bb = Bsm + cur * (128 * LDT);
#pragma unroll
    for (int ks = 0; ks < KSTEP; ks += 32) {
      Frag af[2];
#pragma unroll
      for (int mf = 0; mf < 2; ++mf) {
        const __bf16* p = Ab + (mwbase + mf * 16 + lmod) * LDT + ks + lhalf * 8;
        af[mf].q[0] = *reinterpret_cast<const uint4*>(p);
        af[mf].q[1] = *reinterpret_cast<const uint4*>(p + 16);
      }
#pragma unroll
      for (int j = 0; j < 4; ++j) {
        const __bf16* p = Bb + (nwbase + j * 16 + lmod) * LDT + ks + lhalf * 8;
        Frag bf;
        bf.q[0] = *reinterpret_cast<const uint4*>(p);
        bf.q[1] = *reinterpret_cast<const uint4*>(p + 16);
#pragma unroll
        for (int mf = 0; mf < 2; ++mf)
          acc[mf][j] = __builtin_amdgcn_wmma_f32_16x16x32_bf16(
              false, af[mf].v, false, bf.v, (short)0, acc[mf][j], false, false);
      }
    }
    if (more) {
      const int nb = (cur ^ 1) * (128 * LDT);
#pragma unroll
      for (int c = 0; c < 4; ++c) {
        *reinterpret_cast<uint4*>(Asm + nb + sst + c * 8) = pa[c];
        *reinterpret_cast<uint4*>(Bsm + nb + sst + c * 8) = pb[c];
      }
    }
    __syncthreads();
  }

  // C/D layout: element (m = mwbase + mf*16 + r + 8*lhalf, n = nwbase + j*16 + lmod)
  const int n0 = ntile * 128 + nwbase;
  if constexpr (IS_UP) {
#pragma unroll
    for (int mf = 0; mf < 2; ++mf)
#pragma unroll
      for (int j = 0; j < 4; ++j) {
        int n = n0 + j * 16 + lmod;
        float bv = bias[e * NDIM + n];
#pragma unroll
        for (int r = 0; r < 8; ++r) {
          int m = mwbase + mf * 16 + r + 8 * lhalf;
          if (mtile * 128 + m < cnt) {
            float v = acc[mf][j][r] + bv;
            float g = 0.5f * v * (1.0f + erff(v * 0.70710678118654752f)); // exact gelu
            Hout[(long)asg[m] * UDIM + n] = (__bf16)g;
          }
        }
      }
  } else {
#pragma unroll
    for (int mf = 0; mf < 2; ++mf) {
      float gv[8]; int av[8]; bool ok[8];
#pragma unroll
      for (int r = 0; r < 8; ++r) {
        int m = mwbase + mf * 16 + r + 8 * lhalf;
        ok[r] = (mtile * 128 + m) < cnt;
        av[r] = asg[m];
        gv[r] = gate[av[r]];
      }
#pragma unroll
      for (int j = 0; j < 4; ++j) {
        int n = n0 + j * 16 + lmod;
        float bv = bias[e * NDIM + n];
#pragma unroll
        for (int r = 0; r < 8; ++r)
          if (ok[r]) Dout[(long)av[r] * DDIM + n] = gv[r] * (acc[mf][j][r] + bv);
      }
    }
  }
}

// ---------------- combine: out[t] = row(2t) + row(2t+1) ----------------
__global__ void combine_kernel(const float* __restrict__ dn,
                               float* __restrict__ out, long n4) {
  long i = (long)blockIdx.x * blockDim.x + threadIdx.x;
  long stride = (long)gridDim.x * blockDim.x;
  for (; i < n4; i += stride) {
    long base = i * 4;
    long t = base / DDIM;
    long d = base - t * DDIM;
    const float4 p = *reinterpret_cast<const float4*>(dn + (2 * t) * DDIM + d);
    const float4 q = *reinterpret_cast<const float4*>(dn + (2 * t + 1) * DDIM + d);
    float4 o; o.x = p.x + q.x; o.y = p.y + q.y; o.z = p.z + q.z; o.w = p.w + q.w;
    reinterpret_cast<float4*>(out)[i] = o;
  }
}

extern "C" void kernel_launch(void* const* d_in, const int* in_sizes, int n_in,
                              void* d_out, int out_size, void* d_ws, size_t ws_size,
                              hipStream_t stream) {
  const float* x        = (const float*)d_in[0];
  const float* w_up     = (const float*)d_in[1];
  const float* w_down   = (const float*)d_in[2];
  const float* b_up     = (const float*)d_in[3];
  const float* b_down   = (const float*)d_in[4];
  const float* w_router = (const float*)d_in[5];

  char* ws = (char*)d_ws;
  size_t off = 0;
  auto carve = [&](size_t bytes) {
    size_t o = off;
    off = (off + bytes + 255) & ~(size_t)255;
    return o;
  };
  __bf16* xb   = (__bf16*)(ws + carve((size_t)TOK * DDIM * 2));          // 8 MB
  __bf16* wub  = (__bf16*)(ws + carve((size_t)NEXP * UDIM * DDIM * 2));  // 64 MB
  __bf16* wdb  = (__bf16*)(ws + carve((size_t)NEXP * DDIM * UDIM * 2));  // 64 MB
  __bf16* hb   = (__bf16*)(ws + carve((size_t)NASG * UDIM * 2));         // 64 MB
  float*  dnb  = (float*) (ws + carve((size_t)NASG * DDIM * 4));         // 32 MB
  float*  gate = (float*) (ws + carve((size_t)NASG * 4));
  int*    cnts = (int*)   (ws + carve((size_t)NEXP * 4));
  int*    list = (int*)   (ws + carve((size_t)NEXP * TOK * 4));

  hipMemsetAsync(cnts, 0, NEXP * sizeof(int), stream);

  // fp32 -> bf16 (bf16 weights: 128 MB working set < 192 MB L2)
  {
    long n4x = (long)TOK * DDIM / 4;
    cvt_bf16_kernel<<<(unsigned)((n4x + 255) / 256), 256, 0, stream>>>(x, xb, n4x);
    long n4w = (long)NEXP * UDIM * DDIM / 4;
    cvt_bf16_kernel<<<32768, 256, 0, stream>>>(w_up, wub, n4w);
    cvt_bf16_kernel<<<32768, 256, 0, stream>>>(w_down, wdb, n4w);
  }

  router_kernel<<<TOK / 8, 256, 0, stream>>>(x, w_router, gate, cnts, list);

  // up: [cnt x D] @ [D x U]^T -> gelu -> h
  moe_gemm_kernel<DDIM, true>
      <<<dim3(UDIM / 128, TOK / 128, NEXP), 256, SMEM_BYTES, stream>>>(
          xb, wub, b_up, gate, cnts, list, hb, nullptr);
  // down: [cnt x U] @ [U x D]^T -> *gate -> per-assignment rows
  moe_gemm_kernel<UDIM, false>
      <<<dim3(DDIM / 128, TOK / 128, NEXP), 256, SMEM_BYTES, stream>>>(
          hb, wdb, b_down, gate, cnts, list, nullptr, dnb);

  combine_kernel<<<4096, 256, 0, stream>>>(dnb, (float*)d_out, (long)TOK * DDIM / 4);
}
